// MSAColumnWiseAttention_36060545417653
// MI455X (gfx1250) — compile-verified, matrix operand order checked by reading, not run on hardware
//
#include <hip/hip_runtime.h>
#include <hip/hip_bf16.h>

// MSA column-wise attention for gfx1250 (MI455X), wave32 + WMMA f16 +
// async global->LDS staging. Shapes: B=1, S=R=C=256, H=8, D=32.

typedef __attribute__((ext_vector_type(16))) _Float16 v16h;
typedef __attribute__((ext_vector_type(8)))  _Float16 v8h;
typedef __attribute__((ext_vector_type(8)))  float    v8f;

#define NS 256
#define NR 256
#define NC 256
#define NH 8
#define ND 32
#define LN_EPS 1e-5f

__device__ inline v8f vzero8f() {
  v8f z;
#pragma unroll
  for (int i = 0; i < 8; ++i) z[i] = 0.0f;
  return z;
}

// Load a 16x32 f16 tile in WMMA A-matrix register layout (ISA 7.12.2):
//   lanes 0-15  : row m = lane,    K = {0..7, 16..23}
//   lanes 16-31 : row m = lane-16, K = {8..15, 24..31}
// base points at tile element (0,0); ld = row stride in elements; k0 = K offset.
// Also used for B operands under B-layout == A-layout of B^T (base = B^T).
__device__ inline v16h load_tileA(const _Float16* base, int ld, int k0, int lane) {
  const int m  = lane & 15;
  const int kb = (lane < 16) ? 0 : 8;
  const _Float16* p = base + (size_t)m * ld + k0 + kb;
  v8h lo = *(const v8h*)(p);        // K = kb .. kb+7
  v8h hi = *(const v8h*)(p + 16);   // K = kb+16 .. kb+23
  v16h a;
#pragma unroll
  for (int i = 0; i < 8; ++i) { a[i] = lo[i]; a[i + 8] = hi[i]; }
  return a;
}

__device__ inline v8f wmma_f16(v16h a, v16h b, v8f c) {
  return __builtin_amdgcn_wmma_f32_16x16x32_f16(false, a, false, b, (short)0, c,
                                                false, false);
}

// Async copy of one 16-byte chunk global -> LDS (ASYNCcnt tracked, no VGPR data).
__device__ inline void async_g2l_b128(const _Float16* gsrc, _Float16* lds_dst) {
  unsigned lds_addr = (unsigned)(size_t)lds_dst;              // LDS offset = addr[31:0]
  unsigned long long gaddr = (unsigned long long)(size_t)gsrc;
  asm volatile("global_load_async_to_lds_b128 %0, %1, off"
               :: "v"(lds_addr), "v"(gaddr) : "memory");
}
__device__ inline void wait_asynccnt0() {
  asm volatile("s_wait_asynccnt 0x0" ::: "memory");
}

// ---------------------------------------------------------------------------
// Kernel 1: W[k][n] (f32) -> Wt[n][k] (f16).  grid=256 (n), block=256 (k).
// ---------------------------------------------------------------------------
__global__ __launch_bounds__(256) void k_cvtw(const float* __restrict__ W,
                                              _Float16* __restrict__ Wt) {
  int n = blockIdx.x, k = threadIdx.x;
  Wt[(size_t)n * 256 + k] = (_Float16)W[(size_t)k * 256 + n];
}

// ---------------------------------------------------------------------------
// Kernel 2: LayerNorm over s for each (r,c), emit f16 xh[r][s][c].
// grid = 256 (r), block = 256 (c). Second pass reads hit L2 (input 64MB < 192MB).
// ---------------------------------------------------------------------------
__global__ __launch_bounds__(256) void k_ln(const float* __restrict__ x,
                                            const float* __restrict__ lnw,
                                            const float* __restrict__ lnb,
                                            _Float16* __restrict__ xh) {
  const int r = blockIdx.x, c = threadIdx.x;
  float sum = 0.f, sum2 = 0.f;
  for (int s = 0; s < NS; ++s) {
    float v = x[(size_t)s * (NR * NC) + (size_t)r * NC + c];
    sum += v; sum2 += v * v;
  }
  const float mu   = sum * (1.0f / NS);
  const float var  = sum2 * (1.0f / NS) - mu * mu;
  const float rstd = rsqrtf(var + LN_EPS);
  for (int s = 0; s < NS; ++s) {
    float v = x[(size_t)s * (NR * NC) + (size_t)r * NC + c];
    float y = (v - mu) * rstd * lnw[s] + lnb[s];
    xh[(size_t)r * (NS * NC) + (size_t)s * NC + c] = (_Float16)y;
  }
}

// ---------------------------------------------------------------------------
// Kernel 3: fused Q/K/V/G projections.
// M = R*S = 65536 (m = r*256+s), N = 4*256, K = 256.
// grid = 512, block = 256 (8 waves). Wave w owns m-tile = bx*8+w, loops the
// 64 n-tiles. A sub-tiles (8x8 VGPRs) hoisted; B sub-tiles batch-loaded so
// the 8 WMMAs issue behind a single load wait.
// ---------------------------------------------------------------------------
__global__ __launch_bounds__(256) void k_proj(const _Float16* __restrict__ xh,
                                              const _Float16* __restrict__ wt4, // q,k,v,g each [n][k]
                                              const float* __restrict__ bg,
                                              _Float16* __restrict__ q,
                                              _Float16* __restrict__ kM,
                                              _Float16* __restrict__ vT,
                                              _Float16* __restrict__ g) {
  const int lane = threadIdx.x & 31, w = threadIdx.x >> 5;
  const int mt = blockIdx.x * 8 + w;            // 0..4095
  const int r  = mt >> 4;
  const int s0 = (mt & 15) * 16;
  const _Float16* Ab = xh + (size_t)mt * 16 * 256;

  v16h at[8];
#pragma unroll
  for (int kc = 0; kc < 8; ++kc) at[kc] = load_tileA(Ab, 256, kc * 32, lane);

  for (int nt = 0; nt < 64; ++nt) {
    const int mat = nt >> 4;                    // 0=q 1=k 2=v 3=g
    const int n0  = (nt & 15) * 16;
    const _Float16* Bb = wt4 + ((size_t)mat * 256 + n0) * 256;
    v16h bt[8];
#pragma unroll
    for (int kc = 0; kc < 8; ++kc) bt[kc] = load_tileA(Bb, 256, kc * 32, lane);
    v8f acc = vzero8f();
#pragma unroll
    for (int kc = 0; kc < 8; ++kc) acc = wmma_f16(at[kc], bt[kc], acc);

    const int nl = lane & 15;
    const int n  = n0 + nl;
    const int hh = n >> 5, dd = n & 31;
#pragma unroll
    for (int gi = 0; gi < 8; ++gi) {
      const int ml = gi + ((lane >= 16) ? 8 : 0);   // C-layout row
      const int s  = s0 + ml;
      const float v = acc[gi];
      if (mat == 3) {
        float z = v + bg[n];
        float sg = 1.0f / (1.0f + __expf(-z));
        g[((size_t)(r * 256 + s)) * 256 + n] = (_Float16)sg;
      } else if (mat == 0) {
        q[(((size_t)(r * NH + hh)) * NS + s) * ND + dd] = (_Float16)v;
      } else if (mat == 1) {
        kM[(((size_t)(r * NH + hh)) * NS + s) * ND + dd] = (_Float16)v;
      } else {
        vT[(((size_t)(r * NH + hh)) * ND + dd) * NS + s] = (_Float16)v;
      }
    }
  }
}

// ---------------------------------------------------------------------------
// Kernel 4: attention per (r,h). grid = (2048, 2), block = 256 (8 waves).
// K tile (256x32 f16, 16KB) and V^T tile (32x256 f16, 16KB) are staged ONCE
// per workgroup into LDS via global_load_async_to_lds_b128 (shared by all 8
// waves -> 8x less global traffic), then consumed as WMMA B operands.
// Softmax: shfl_xor {8,4,2,1} reduces the 16-lane halves (C/D row layout).
// P goes through LDS to convert C-layout -> A-layout for O = P.V.
// ---------------------------------------------------------------------------
__global__ __launch_bounds__(256) void k_attn(const _Float16* __restrict__ q,
                                              const _Float16* __restrict__ kM,
                                              const _Float16* __restrict__ vT,
                                              const _Float16* __restrict__ g,
                                              _Float16* __restrict__ og) {
  __shared__ _Float16 Ksh[NS * ND];             // 16 KB  [s][d]
  __shared__ _Float16 Vsh[ND * NS];             // 16 KB  [d][s] (V^T)
  __shared__ _Float16 P[8][16][256];            // 64 KB  per-wave P tiles
  const int tid  = threadIdx.x;
  const int lane = tid & 31, w = tid >> 5;
  const int rh = blockIdx.x;                    // r*8 + h
  const int r = rh >> 3, h = rh & 7;
  const int i0 = blockIdx.y * 128 + w * 16;

  // ---- async stage K and V^T tiles into LDS (1024 x b128 chunks each) ----
  {
    const _Float16* kg = kM + (size_t)rh * NS * ND;
    const _Float16* vg = vT + (size_t)rh * ND * NS;
#pragma unroll
    for (int it = 0; it < 4; ++it) {
      const int elt = (tid + it * 256) * 8;     // 8 halves per chunk
      async_g2l_b128(kg + elt, &Ksh[elt]);
      async_g2l_b128(vg + elt, &Vsh[elt]);
    }
    wait_asynccnt0();
  }
  __syncthreads();

  // ---- logits: S = Q K^T, one WMMA per 16-col j-tile (K-dim = d = 32) ----
  const _Float16* qb = q + ((size_t)rh * NS + i0) * ND;
  const v16h aq = load_tileA(qb, ND, 0, lane);  // 16 query rows, loaded once
  v8f acc[16];
#pragma unroll
  for (int jt = 0; jt < 16; ++jt) acc[jt] = vzero8f();
#pragma unroll
  for (int jt = 0; jt < 16; jt += 4) {
    v16h b0 = load_tileA(&Ksh[(jt + 0) * 16 * ND], ND, 0, lane);
    v16h b1 = load_tileA(&Ksh[(jt + 1) * 16 * ND], ND, 0, lane);
    v16h b2 = load_tileA(&Ksh[(jt + 2) * 16 * ND], ND, 0, lane);
    v16h b3 = load_tileA(&Ksh[(jt + 3) * 16 * ND], ND, 0, lane);
    acc[jt + 0] = wmma_f16(aq, b0, acc[jt + 0]);
    acc[jt + 1] = wmma_f16(aq, b1, acc[jt + 1]);
    acc[jt + 2] = wmma_f16(aq, b2, acc[jt + 2]);
    acc[jt + 3] = wmma_f16(aq, b3, acc[jt + 3]);
  }

  // ---- row softmax (rows live in 16-lane halves of the C layout) ----
  const float scale = 0.17677669529663687f;     // 1/sqrt(32)
#pragma unroll
  for (int gi = 0; gi < 8; ++gi) {
    float mx = -3.0e38f;
#pragma unroll
    for (int jt = 0; jt < 16; ++jt) mx = fmaxf(mx, acc[jt][gi]);
#pragma unroll
    for (int m = 8; m >= 1; m >>= 1) mx = fmaxf(mx, __shfl_xor(mx, m, 32));
    float sum = 0.f;
#pragma unroll
    for (int jt = 0; jt < 16; ++jt) {
      float e = __expf((acc[jt][gi] - mx) * scale);
      acc[jt][gi] = e; sum += e;
    }
#pragma unroll
    for (int m = 8; m >= 1; m >>= 1) sum += __shfl_xor(sum, m, 32);
    const float inv = 1.0f / sum;
    const int ml = gi + ((lane >= 16) ? 8 : 0);
    const int nl = lane & 15;
#pragma unroll
    for (int jt = 0; jt < 16; ++jt)
      P[w][ml][jt * 16 + nl] = (_Float16)(acc[jt][gi] * inv);
  }
  __syncthreads();

  // ---- O = P.V : K-dim = j in chunks of 32, N = d in two 16-wide tiles ----
  v8f o0 = vzero8f(), o1 = vzero8f();
#pragma unroll
  for (int kc = 0; kc < 8; ++kc) {
    v16h ap = load_tileA(&P[w][0][0], 256, kc * 32, lane);
    v16h b0 = load_tileA(&Vsh[0],        NS, kc * 32, lane); // d = 0..15
    v16h b1 = load_tileA(&Vsh[16 * NS],  NS, kc * 32, lane); // d = 16..31
    o0 = wmma_f16(ap, b0, o0);
    o1 = wmma_f16(ap, b1, o1);
  }

  // ---- gate + store f16 OG[r][s][h*32+d] ----
  const int nl = lane & 15;
#pragma unroll
  for (int gi = 0; gi < 8; ++gi) {
    const int ml = gi + ((lane >= 16) ? 8 : 0);
    const int s  = i0 + ml;
    const size_t rowoff = ((size_t)(r * 256 + s)) * 256 + h * 32;
    float g0 = (float)g[rowoff + nl];
    float g1 = (float)g[rowoff + 16 + nl];
    og[rowoff + nl]      = (_Float16)(o0[gi] * g0);
    og[rowoff + 16 + nl] = (_Float16)(o1[gi] * g1);
  }
}

// ---------------------------------------------------------------------------
// Kernel 5: output projection. out = OG @ Wo + bo, stored transposed [s][r][c].
// grid = 512, block = 256. Wave owns an m-tile, loops 16 n-tiles; batched
// B loads ahead of the WMMA run.
// ---------------------------------------------------------------------------
__global__ __launch_bounds__(256) void k_out(const _Float16* __restrict__ og,
                                             const _Float16* __restrict__ wto, // [n][k]
                                             const float* __restrict__ bo,
                                             float* __restrict__ out) {
  const int lane = threadIdx.x & 31, w = threadIdx.x >> 5;
  const int mt = blockIdx.x * 8 + w;            // 0..4095
  const int r  = mt >> 4;
  const int s0 = (mt & 15) * 16;
  const _Float16* Ab = og + (size_t)mt * 16 * 256;

  v16h at[8];
#pragma unroll
  for (int kc = 0; kc < 8; ++kc) at[kc] = load_tileA(Ab, 256, kc * 32, lane);

  for (int nt = 0; nt < 16; ++nt) {
    const _Float16* Bb = wto + (size_t)nt * 16 * 256;
    v16h bt[8];
#pragma unroll
    for (int kc = 0; kc < 8; ++kc) bt[kc] = load_tileA(Bb, 256, kc * 32, lane);
    v8f acc = vzero8f();
#pragma unroll
    for (int kc = 0; kc < 8; ++kc) acc = wmma_f16(at[kc], bt[kc], acc);

    const int n = nt * 16 + (lane & 15);
    const float bias = bo[n];
#pragma unroll
    for (int gi = 0; gi < 8; ++gi) {
      const int s = s0 + gi + ((lane >= 16) ? 8 : 0);
      out[((size_t)s * NR + r) * NC + n] = acc[gi] + bias;
    }
  }
}

// ---------------------------------------------------------------------------
extern "C" void kernel_launch(void* const* d_in, const int* in_sizes, int n_in,
                              void* d_out, int out_size, void* d_ws, size_t ws_size,
                              hipStream_t stream) {
  const float* x   = (const float*)d_in[0];  // msa_repr [1,S,R,C]
  const float* lnw = (const float*)d_in[1];
  const float* lnb = (const float*)d_in[2];
  const float* Wq  = (const float*)d_in[3];
  const float* Wk  = (const float*)d_in[4];
  const float* Wv  = (const float*)d_in[5];
  const float* Wg  = (const float*)d_in[6];
  const float* bg  = (const float*)d_in[7];
  const float* Wo  = (const float*)d_in[8];
  const float* bo  = (const float*)d_in[9];
  float* out = (float*)d_out;

  char* ws = (char*)d_ws;
  size_t off = 0;
  const size_t big = (size_t)NR * NS * 256 * sizeof(_Float16);   // 32 MB
  _Float16* xh  = (_Float16*)(ws + off); off += big;
  _Float16* wt4 = (_Float16*)(ws + off); off += (size_t)4 * 256 * 256 * sizeof(_Float16);
  _Float16* wto = (_Float16*)(ws + off); off += (size_t)256 * 256 * sizeof(_Float16);
  _Float16* qB  = (_Float16*)(ws + off); off += big;
  _Float16* kB  = (_Float16*)(ws + off); off += big;
  _Float16* vTB = (_Float16*)(ws + off); off += big;
  _Float16* gB  = (_Float16*)(ws + off); off += big;
  _Float16* ogB = (_Float16*)(ws + off); off += big;

  // 1. weights -> f16, transposed to [n][k]
  k_cvtw<<<256, 256, 0, stream>>>(Wq, wt4 + 0 * 65536);
  k_cvtw<<<256, 256, 0, stream>>>(Wk, wt4 + 1 * 65536);
  k_cvtw<<<256, 256, 0, stream>>>(Wv, wt4 + 2 * 65536);
  k_cvtw<<<256, 256, 0, stream>>>(Wg, wt4 + 3 * 65536);
  k_cvtw<<<256, 256, 0, stream>>>(Wo, wto);

  // 2. LayerNorm over s -> f16 xh[r][s][c]
  k_ln<<<NR, NC, 0, stream>>>(x, lnw, lnb, xh);

  // 3. Q/K/V/G projections (WMMA)
  k_proj<<<512, 256, 0, stream>>>(xh, wt4, bg, qB, kB, vTB, gB);

  // 4. attention per (r,h): async LDS staging + WMMA + wave32 softmax
  k_attn<<<dim3(NR * NH, 2), 256, 0, stream>>>(qB, kB, vTB, gB, ogB);

  // 5. output projection + bias, transpose back (WMMA)
  k_out<<<512, 256, 0, stream>>>(ogB, wto, bo, out);
}
